// CorrFusion_49228915147505
// MI455X (gfx1250) — matrix operand order repaired and checked by reading, not compiled, not run on hardware
//
#include <hip/hip_runtime.h>
#include <hip/hip_bf16.h>

// Drop FP canonicalization / strict NaN semantics: removes the
// v_max_num_f32 x,x,x self-canonicalize ops LLVM inserts before maxnum.
#pragma float_control(precise, off)

typedef __attribute__((ext_vector_type(16))) _Float16 v16h;
typedef __attribute__((ext_vector_type(8)))  _Float16 v8h;
typedef __attribute__((ext_vector_type(8)))  float    v8f;

#define N_PTS 1024
#define NCHUNK 8
#define ICHUNK (N_PTS / NCHUNK)   // 128 -> 8 i-tiles of 16 per wave

__device__ __forceinline__ v16h cat8(v8h lo, v8h hi) {
    return __builtin_shufflevector(lo, hi, 0,1,2,3,4,5,6,7,8,9,10,11,12,13,14,15);
}

// ---------------- init: zero the running-max buffer ----------------
__global__ void k_init(unsigned* __restrict__ p, int n) {
    int i = blockIdx.x * blockDim.x + threadIdx.x;
    if (i < n) p[i] = 0u;   // relu'd max is >= 0, so 0 == -inf here
}

// ---------------- P1: A[i,k] = f1[i]@W1a^T ; B[j,k] = f2[j]@W1b^T + b1_0 (to f16) ----
__global__ void k_prepAB(const float* __restrict__ pcf1, const float* __restrict__ pcf2,
                         const float* __restrict__ W10, const float* __restrict__ b10,
                         _Float16* __restrict__ Ah, _Float16* __restrict__ Bh) {
    int gid = blockIdx.x * blockDim.x + threadIdx.x;     // 0 .. 131071
    int which = gid >> 16;            // 0 = A, 1 = B
    int i = (gid >> 6) & 1023;
    int k = gid & 63;
    const float* feat = which ? pcf2 : pcf1;             // [64][1024]
    const float* wrow = W10 + k * 128 + (which ? 64 : 0);
    float s = which ? b10[k] : 0.0f;
    #pragma unroll 8
    for (int c = 0; c < 64; ++c) s += feat[c * N_PTS + i] * wrow[c];
    (which ? Bh : Ah)[i * 64 + k] = (_Float16)s;
}

// ---------------- P2: W1_1 -> f16 ----------------
__global__ void k_prepW(const float* __restrict__ W11, _Float16* __restrict__ Wh) {
    int i = blockIdx.x * blockDim.x + threadIdx.x;
    if (i < 64 * 64) Wh[i] = (_Float16)W11[i];
}

// ---------------- main fused WMMA kernel ----------------
// per (i,j): h = relu(A_i + B_j); o = relu(h @ W1_1^T + b1_1); max over i -> hmax[j, 64]
// max is deferred: elementwise v8f running max of raw WMMA tiles, one reduction per wave.
__global__ __launch_bounds__(256) void k_pairmax(
    const _Float16* __restrict__ Ah,   // [1024][64]
    const _Float16* __restrict__ Bh,   // [1024][64] (includes b1_0)
    const _Float16* __restrict__ Wh,   // [64][64]  W1_1 row-major [o][c]
    const float*    __restrict__ b11,  // [64]
    unsigned*       __restrict__ hmaxU // [1024][64] float-bits, init 0
) {
    const int lane  = threadIdx.x & 31;
    const int wave  = threadIdx.x >> 5;       // 0..7
    const int jb    = blockIdx.x & 127;
    const int chunk = blockIdx.x >> 7;        // 0..NCHUNK-1
    const int j     = jb * 8 + wave;
    const int i0c   = chunk * ICHUNK;

    const int nsub = lane & 15;               // M index (A rows) or N index (B/W/C cols)
    const int hi   = lane >> 4;

    // W1_1 as WMMA B-operands, same K-swizzle as A: t=0..7 -> K = t + 8*hi ; t=8..15 -> K = (t-8) + 16 + 8*hi
    v16h w0[4], w1[4];
    float bias[4];
    #pragma unroll
    for (int nt = 0; nt < 4; ++nt) {
        const v8h* pw = (const v8h*)(Wh + (nt * 16 + nsub) * 64);
        w0[nt] = cat8(pw[hi],     pw[2 + hi]);   // K = 8*hi.., 16+8*hi..
        w1[nt] = cat8(pw[4 + hi], pw[6 + hi]);   // K = 32+8*hi.., 48+8*hi..
        bias[nt] = b11[nt * 16 + nsub];
    }

    // B_j broadcast term, in the A-matrix K-swizzle:
    // t=0..7 -> K = t + 8*hi ; t=8..15 -> K = (t-8) + 16 + 8*hi  (then +32 for 2nd half)
    const v8h* pb = (const v8h*)(Bh + j * 64);
    v16h bj0 = cat8(pb[hi],     pb[2 + hi]);
    v16h bj1 = cat8(pb[4 + hi], pb[6 + hi]);

    const v16h z16 = {};
    v8f runC[4];
    #pragma unroll
    for (int nt = 0; nt < 4; ++nt)
        #pragma unroll
        for (int e = 0; e < 8; ++e) runC[nt][e] = -3.4e38f;

    for (int it = 0; it < ICHUNK / 16; ++it) {
        const int row = i0c + it * 16 + nsub;
        const v8h* pa = (const v8h*)(Ah + row * 64);
        __builtin_prefetch((const void*)(Ah + (row + 16) * 64), 0, 3);
        v16h a0 = cat8(pa[hi],     pa[2 + hi]);
        v16h a1 = cat8(pa[4 + hi], pa[6 + hi]);
        v16h h0 = __builtin_elementwise_max(a0 + bj0, z16);  // relu(A_i + B_j)
        v16h h1 = __builtin_elementwise_max(a1 + bj1, z16);

        #pragma unroll
        for (int nt = 0; nt < 4; ++nt) {
            v8f c = {};
            c = __builtin_amdgcn_wmma_f32_16x16x32_f16(false, h0, false, w0[nt],
                                                       (short)0, c, false, false);
            c = __builtin_amdgcn_wmma_f32_16x16x32_f16(false, h1, false, w1[nt],
                                                       (short)0, c, false, false);
            runC[nt] = __builtin_elementwise_max(runC[nt], c);  // defer reduction
        }
    }

    // one reduction per wave: over 8 M-rows in regs, then cross-half, +bias, relu
    #pragma unroll
    for (int nt = 0; nt < 4; ++nt) {
        float tm = runC[nt][0];
        #pragma unroll
        for (int e = 1; e < 8; ++e) tm = fmaxf(tm, runC[nt][e]);
        tm = fmaxf(tm, __shfl_xor(tm, 16, 32));
        float val = fmaxf(bias[nt] + tm, 0.0f);
        if (lane < 16) {
            // val >= 0, target init 0 -> uint ordering == float ordering
            atomicMax(&hmaxU[j * 64 + nt * 16 + lane], __float_as_uint(val));
        }
    }
}

// ---------------- T1: m = relu(hmax @ W2_0^T + b2_0) ----------------
__global__ void k_t1(const float* __restrict__ Hm, const float* __restrict__ W20,
                     const float* __restrict__ b20, float* __restrict__ M) {
    int gid = blockIdx.x * blockDim.x + threadIdx.x;  // 65536
    int n = gid >> 6, o = gid & 63;
    const float* h = Hm + n * 64;
    const float* w = W20 + o * 64;
    float s = b20[o];
    #pragma unroll 8
    for (int k = 0; k < 64; ++k) s += h[k] * w[k];
    M[n * 64 + o] = fmaxf(s, 0.0f);
}

// ---------------- T2: feat_flow (+transpose out) and x = pc_1 + flow ----------------
__global__ void k_t2(const float* __restrict__ M, const float* __restrict__ W21,
                     const float* __restrict__ b21, const float* __restrict__ pc1,
                     float* __restrict__ out0, float* __restrict__ X) {
    int n = blockIdx.x * blockDim.x + threadIdx.x;
    if (n >= N_PTS) return;
    const float* m = M + n * 64;
    #pragma unroll
    for (int c = 0; c < 3; ++c) {
        const float* w = W21 + c * 64;
        float s = b21[c];
        #pragma unroll 8
        for (int k = 0; k < 64; ++k) s += m[k] * w[k];
        out0[c * N_PTS + n] = s;                 // [3,1024] transposed output
        X[n * 3 + c] = pc1[n * 3 + c] + s;
    }
}

// ---------------- T3: pointfeat = relu(x @ Wp1^T + bp1); also -> out1[:,1024:] ----------
__global__ void k_t3(const float* __restrict__ X, const float* __restrict__ Wp1,
                     const float* __restrict__ bp1, float* __restrict__ PF,
                     float* __restrict__ out1) {
    int gid = blockIdx.x * blockDim.x + threadIdx.x;  // 65536
    int n = gid >> 6, o = gid & 63;
    float s = bp1[o];
    #pragma unroll
    for (int c = 0; c < 3; ++c) s += X[n * 3 + c] * Wp1[o * 3 + c];
    s = fmaxf(s, 0.0f);
    PF[n * 64 + o] = s;
    out1[n * 1088 + 1024 + o] = s;
}

// ---------------- T4: h2 = relu(pf @ Wp2^T + bp2) ----------------
__global__ void k_t4(const float* __restrict__ PF, const float* __restrict__ Wp2,
                     const float* __restrict__ bp2, float* __restrict__ H2) {
    int gid = blockIdx.x * blockDim.x + threadIdx.x;  // 131072
    int n = gid >> 7, o = gid & 127;
    const float* p = PF + n * 64;
    const float* w = Wp2 + o * 64;
    float s = bp2[o];
    #pragma unroll 8
    for (int k = 0; k < 64; ++k) s += p[k] * w[k];
    H2[n * 128 + o] = fmaxf(s, 0.0f);
}

// ---------------- T5: gfeat[o] = max_n (h2[n] . Wp3[o]) + bp3[o] ----------------
__global__ __launch_bounds__(128) void k_t5(const float* __restrict__ H2,
                                            const float* __restrict__ Wp3,
                                            const float* __restrict__ bp3,
                                            float* __restrict__ G) {
    const int o = blockIdx.x;
    const float* w = Wp3 + o * 128;
    float mx = -3.4e38f;
    for (int n = threadIdx.x; n < N_PTS; n += 128) {
        const float* h = H2 + n * 128;
        float s = 0.0f;
        #pragma unroll 8
        for (int c = 0; c < 128; ++c) s += h[c] * w[c];
        mx = fmaxf(mx, s);
    }
    __shared__ float red[128];
    red[threadIdx.x] = mx;
    __syncthreads();
    for (int s = 64; s > 0; s >>= 1) {
        if (threadIdx.x < s) red[threadIdx.x] = fmaxf(red[threadIdx.x], red[threadIdx.x + s]);
        __syncthreads();
    }
    if (threadIdx.x == 0) G[o] = red[0] + bp3[o];
}

// ---------------- T6: broadcast gfeat into out1[:, 0:1024] ----------------
__global__ void k_t6(const float* __restrict__ G, float* __restrict__ out1) {
    int gid = blockIdx.x * blockDim.x + threadIdx.x;   // 1M
    int n = gid >> 10, o = gid & 1023;
    out1[n * 1088 + o] = G[o];
}

extern "C" void kernel_launch(void* const* d_in, const int* in_sizes, int n_in,
                              void* d_out, int out_size, void* d_ws, size_t ws_size,
                              hipStream_t stream) {
    const float* pc_1  = (const float*)d_in[0];
    const float* pcf1  = (const float*)d_in[2];
    const float* pcf2  = (const float*)d_in[3];
    const float* W1_0  = (const float*)d_in[4];
    const float* b1_0  = (const float*)d_in[5];
    const float* W1_1  = (const float*)d_in[6];
    const float* b1_1  = (const float*)d_in[7];
    const float* W2_0  = (const float*)d_in[8];
    const float* b2_0  = (const float*)d_in[9];
    const float* W2_1  = (const float*)d_in[10];
    const float* b2_1  = (const float*)d_in[11];
    const float* Wp1   = (const float*)d_in[12];
    const float* bp1   = (const float*)d_in[13];
    const float* Wp2   = (const float*)d_in[14];
    const float* bp2   = (const float*)d_in[15];
    const float* Wp3   = (const float*)d_in[16];
    const float* bp3   = (const float*)d_in[17];

    char* ws = (char*)d_ws;
    _Float16* Ah   = (_Float16*)(ws + 0);            // 128 KB
    _Float16* Bh   = (_Float16*)(ws + 131072);       // 128 KB
    _Float16* Wh   = (_Float16*)(ws + 262144);       //   8 KB
    float*    Hm   = (float*)   (ws + 270336);       // 256 KB  (hmax, float bits)
    float*    M    = (float*)   (ws + 532480);       // 256 KB
    float*    X    = (float*)   (ws + 794624);       //  12 KB
    float*    PF   = (float*)   (ws + 806912);       // 256 KB
    float*    H2   = (float*)   (ws + 1069056);      // 512 KB
    float*    G    = (float*)   (ws + 1593344);      //   4 KB

    float* out0 = (float*)d_out;          // [3,1024]
    float* out1 = out0 + 3 * N_PTS;       // [1024,1088]

    k_init   <<<256, 256, 0, stream>>>((unsigned*)Hm, N_PTS * 64);
    k_prepAB <<<512, 256, 0, stream>>>(pcf1, pcf2, W1_0, b1_0, Ah, Bh);
    k_prepW  <<<16,  256, 0, stream>>>(W1_1, Wh);
    k_pairmax<<<128 * NCHUNK, 256, 0, stream>>>(Ah, Bh, Wh, b1_1, (unsigned*)Hm);
    k_t1     <<<256, 256, 0, stream>>>(Hm, W2_0, b2_0, M);
    k_t2     <<<4,   256, 0, stream>>>(M, W2_1, b2_1, pc_1, out0, X);
    k_t3     <<<256, 256, 0, stream>>>(X, Wp1, bp1, PF, out1);
    k_t4     <<<512, 256, 0, stream>>>(PF, Wp2, bp2, H2);
    k_t5     <<<1024,128, 0, stream>>>(H2, Wp3, bp3, G);
    k_t6     <<<4096,256, 0, stream>>>(G, out1);
}
